// WSOVODRPN_68083821576583
// MI455X (gfx1250) — matrix operand order repaired and checked by reading, not compiled, not run on hardware
//
#include <hip/hip_runtime.h>
#include <stdint.h>

// ---------------- CDNA5 WMMA types ----------------
typedef __attribute__((ext_vector_type(16))) __bf16 v16bf;
typedef __attribute__((ext_vector_type(8)))  float  v8f;

#define CATN 4256
#define POST_NMS 1000
#define NEGV (-1e30f)

// gfx1250 async global->LDS path (guarded: falls back to sync copy if the
// toolchain doesn't declare the builtins)
#if defined(__has_builtin)
#if __has_builtin(__builtin_amdgcn_global_load_async_to_lds_b128) && \
    __has_builtin(__builtin_amdgcn_s_wait_asynccnt)
#define HAVE_ASYNC_LDS 1
#endif
#endif

#if defined(HAVE_ASYNC_LDS)
// builtin prototype (from hipcc diagnostic): param0 = int4 (vector_size) in AS1
typedef int v4i_t __attribute__((__vector_size__(16)));
typedef __attribute__((address_space(1))) v4i_t as1_v4i;
typedef __attribute__((address_space(3))) v4i_t as3_v4i;
#endif

__device__ __forceinline__ unsigned short f2bf(float f) {
    union { float f; unsigned u; } x; x.f = f;
    unsigned u = x.u;
    unsigned r = u + 0x7fffu + ((u >> 16) & 1u);   // round-to-nearest-even
    return (unsigned short)(r >> 16);
}

// ---------------- Stage 0a: NCHW f32 -> NHWC bf16 ----------------
__global__ void convert_nhwc(const float* __restrict__ in,
                             unsigned short* __restrict__ out,
                             int HW, long n) {
    long gid = (long)blockIdx.x * 256 + threadIdx.x;
    if (gid >= n) return;
    int  c   = (int)(gid & 255);
    long pix = (gid >> 8) % HW;
    long b   = gid / ((long)HW * 256);
    float v = in[(b * 256 + c) * (long)HW + pix];
    out[gid] = f2bf(v);
}

// ---------------- Stage 0b: pack conv weights into WMMA B fragments ----------
// Fragment f = step*16 + nt, step = (ky*3+kx)*8 + kstep.
// Within fragment: lane L, element i -> K = (L/16)*16 + i, N = L%16.
// cin = kstep*32 + (L/16)*16 + i ; cout = nt*16 + L%16.
__global__ void pack_weights(const float* __restrict__ w,   // [256][256][3][3]
                             unsigned short* __restrict__ wp) {
    long gid = (long)blockIdx.x * 256 + threadIdx.x;
    if (gid >= 72l * 16 * 512) return;
    int f = (int)(gid >> 9);
    int e = (int)(gid & 511);
    int L = e >> 4, i = e & 15;
    int nt = f & 15, step = f >> 4;
    int kstep = step & 7, t = step >> 3;
    int kx = t % 3, ky = t / 3;
    int cout = nt * 16 + (L & 15);
    int cin  = kstep * 32 + (L >> 4) * 16 + i;
    float v = w[(((long)cout * 256 + cin) * 3 + ky) * 3 + kx];
    wp[gid] = f2bf(v);
}

// ---------------- Stage 1: fused 3x3 conv + ReLU + 1x1 heads ----------------
union Frag { uint4 u[2]; v16bf v; };

__device__ __forceinline__ void load_afrag(Frag& f, const unsigned short* inb,
                                           int step, int py, int px,
                                           int H, int W, int hh) {
    const int ky = step / 24;
    const int rem = step % 24;
    const int kx = rem / 8;
    const int kstep = rem % 8;
    const int ys = py + ky - 1;
    const int xs = px + kx - 1;
    const bool valid = (ys >= 0) & (ys < H) & (xs >= 0) & (xs < W);
    if (valid) {
        const unsigned short* ap =
            inb + ((long)ys * W + xs) * 256 + kstep * 32 + hh * 16;
        f.u[0] = *(const uint4*)ap;
        f.u[1] = *(const uint4*)(ap + 8);
    } else {
        f.u[0] = make_uint4(0, 0, 0, 0);
        f.u[1] = make_uint4(0, 0, 0, 0);
    }
}

// Grid: x = HW/128 (8 M-tiles of 16 pixels per block), y = batch. 256 thr = 8 waves.
__global__ __launch_bounds__(256)
void conv_rpn_head(const unsigned short* __restrict__ inx,   // [B][HW][256] bf16
                   const unsigned short* __restrict__ wpack, // packed fragments
                   const float* __restrict__ conv_b,
                   const float* __restrict__ obj_w, const float* __restrict__ obj_b,
                   const float* __restrict__ delta_w, const float* __restrict__ delta_b,
                   float* __restrict__ logits,               // [B][HW]
                   float* __restrict__ deltas,               // [B][HW][4]
                   int H, int W) {
    __shared__ unsigned short ldsb[2][8192];   // 2 x 16KB: double-buffered B step

    const int tid  = threadIdx.x;
    const int lane = tid & 31;
    const int ln16 = lane & 15;
    const int hh   = lane >> 4;
    const int b    = blockIdx.y;
    const long HW  = (long)H * W;

    const int  mtile = blockIdx.x * 8 + (tid >> 5);
    const int  p0    = mtile * 16;
    const int  pm    = p0 + ln16;          // A-matrix row M = lane%16
    const int  py    = pm / W;
    const int  px    = pm % W;

    const unsigned short* inb = inx + (long)b * HW * 256;

    v8f acc[16];
    {
        v8f z = {0.f, 0.f, 0.f, 0.f, 0.f, 0.f, 0.f, 0.f};
        #pragma unroll
        for (int nt = 0; nt < 16; ++nt) acc[nt] = z;
    }

    // --- stage helper: copy one 16KB B-step into buffer `buf` (64B/thread) ---
    auto stage = [&](int step, int buf) {
#if defined(HAVE_ASYNC_LDS)
        unsigned short* gsrc =
            (unsigned short*)wpack + (long)step * 8192 + tid * 32;
        unsigned short* ldst = &ldsb[buf][tid * 32];
        __builtin_amdgcn_global_load_async_to_lds_b128(
            (as1_v4i*)gsrc, (as3_v4i*)ldst, 0, 0);
        __builtin_amdgcn_global_load_async_to_lds_b128(
            (as1_v4i*)gsrc, (as3_v4i*)ldst, 16, 0);
        __builtin_amdgcn_global_load_async_to_lds_b128(
            (as1_v4i*)gsrc, (as3_v4i*)ldst, 32, 0);
        __builtin_amdgcn_global_load_async_to_lds_b128(
            (as1_v4i*)gsrc, (as3_v4i*)ldst, 48, 0);
#else
        const uint4* src = (const uint4*)(wpack + (long)step * 8192) + tid * 4;
        uint4*       dst = (uint4*)(&ldsb[buf][0]) + tid * 4;
        #pragma unroll
        for (int q = 0; q < 4; ++q) dst[q] = src[q];
#endif
    };

    // prologue: buffer 0 holds step 0; A fragment for step 0 in flight
    stage(0, 0);
    Frag afr;
    load_afrag(afr, inb, 0, py, px, H, W, hh);
#if defined(HAVE_ASYNC_LDS)
    __builtin_amdgcn_s_wait_asynccnt(0);
#endif
    __syncthreads();

    for (int step = 0; step < 72; ++step) {
        const int cur = step & 1;

        // issue next step's B copy + A load early; they complete under the WMMAs
        if (step < 71) stage(step + 1, cur ^ 1);
        Frag anext;
        if (step < 71) {
            load_afrag(anext, inb, step + 1, py, px, H, W, hh);
        } else {
            anext.u[0] = make_uint4(0, 0, 0, 0);
            anext.u[1] = make_uint4(0, 0, 0, 0);
        }

        // 16 WMMAs on buffer `cur`, B fragments double-buffered in registers
        Frag bcur, bnxt;
        {
            const uint4* bp = (const uint4*)(&ldsb[cur][0] + lane * 16);
            bcur.u[0] = bp[0];
            bcur.u[1] = bp[1];
        }
        #pragma unroll
        for (int nt = 0; nt < 16; ++nt) {
            if (nt < 15) {
                const uint4* bp =
                    (const uint4*)(&ldsb[cur][(nt + 1) * 512] + lane * 16);
                bnxt.u[0] = bp[0];
                bnxt.u[1] = bp[1];
            }
            acc[nt] = __builtin_amdgcn_wmma_f32_16x16x32_bf16(
                false, afr.v, false, bcur.v, (short)0, acc[nt], false, false);
            bcur = bnxt;
        }

#if defined(HAVE_ASYNC_LDS)
        __builtin_amdgcn_s_wait_asynccnt(0);
#endif
        __syncthreads();
        afr = anext;
    }

    // ---- fused heads: bias + ReLU + 1x1 dots, all in registers ----
    float so[8], s0[8], s1[8], s2[8], s3[8];
    #pragma unroll
    for (int r = 0; r < 8; ++r) { so[r]=0.f; s0[r]=0.f; s1[r]=0.f; s2[r]=0.f; s3[r]=0.f; }

    #pragma unroll
    for (int nt = 0; nt < 16; ++nt) {
        const int co = nt * 16 + ln16;
        const float cb = conv_b[co];
        const float wo = obj_w[co];
        const float w0 = delta_w[0 * 256 + co];
        const float w1 = delta_w[1 * 256 + co];
        const float w2 = delta_w[2 * 256 + co];
        const float w3 = delta_w[3 * 256 + co];
        #pragma unroll
        for (int r = 0; r < 8; ++r) {
            float t = acc[nt][r] + cb;
            t = t > 0.f ? t : 0.f;
            so[r] += t * wo;
            s0[r] += t * w0; s1[r] += t * w1; s2[r] += t * w2; s3[r] += t * w3;
        }
    }

    // reduce over the 16 N-lanes of each half-wave (wave32 shuffles)
    #pragma unroll
    for (int m = 1; m < 16; m <<= 1) {
        #pragma unroll
        for (int r = 0; r < 8; ++r) {
            so[r] += __shfl_xor(so[r], m, 32);
            s0[r] += __shfl_xor(s0[r], m, 32);
            s1[r] += __shfl_xor(s1[r], m, 32);
            s2[r] += __shfl_xor(s2[r], m, 32);
            s3[r] += __shfl_xor(s3[r], m, 32);
        }
    }

    if (ln16 == 0) {
        const float ob  = obj_b[0];
        const float d0b = delta_b[0], d1b = delta_b[1], d2b = delta_b[2], d3b = delta_b[3];
        #pragma unroll
        for (int r = 0; r < 8; ++r) {
            const long p = p0 + r + hh * 8;   // D-layout: VGPR r -> M = r (+8 upper half)
            logits[(long)b * HW + p] = so[r] + ob;
            float* dp = deltas + ((long)b * HW + p) * 4;
            dp[0] = s0[r] + d0b; dp[1] = s1[r] + d1b;
            dp[2] = s2[r] + d2b; dp[3] = s3[r] + d3b;
        }
    }
}

// ---------------- Stage 2: exact rank top-k + anchor decode ----------------
// rank(i) = #{j : s_j > s_i or (s_j == s_i and j < i)}  -> stable sorted slot.
__global__ void topk_decode(const float* __restrict__ logits,
                            const float* __restrict__ deltas,
                            float* __restrict__ boxes,   // [B][CATN][4]
                            float* __restrict__ nmsb,    // [B][CATN][4]
                            float* __restrict__ scores,  // [B][CATN]
                            int H, int W, int stride, int k, int lvl_off,
                            float lvl_shift) {
    __shared__ float sc[256];
    const int b = blockIdx.y;
    const int N = H * W;
    const int i = blockIdx.x * 256 + threadIdx.x;
    const float* lg = logits + (long)b * N;
    const float si = lg[i];

    int rank = 0;
    for (int base = 0; base < N; base += 256) {
        sc[threadIdx.x] = lg[base + threadIdx.x];
        __syncthreads();
        #pragma unroll 8
        for (int t = 0; t < 256; ++t) {
            const float sj = sc[t];
            const int   j  = base + t;
            rank += (sj > si) || (sj == si && j < i);
        }
        __syncthreads();
    }

    if (rank < k) {
        const float cx = (float)(i % W) * (float)stride;
        const float cy = (float)(i / W) * (float)stride;
        const float* d = deltas + ((long)b * N + i) * 4;
        float x0 = cx - d[0] * stride, y0 = cy - d[1] * stride;
        float x1 = cx + d[2] * stride, y1 = cy + d[3] * stride;
        x0 = fminf(fmaxf(x0, 0.f), 1024.f);
        y0 = fminf(fmaxf(y0, 0.f), 1024.f);
        x1 = fminf(fmaxf(x1, 0.f), 1024.f);
        y1 = fminf(fmaxf(y1, 0.f), 1024.f);
        const long n = (long)b * CATN + lvl_off + rank;
        boxes[n * 4 + 0] = x0; boxes[n * 4 + 1] = y0;
        boxes[n * 4 + 2] = x1; boxes[n * 4 + 3] = y1;
        nmsb[n * 4 + 0] = x0 + lvl_shift; nmsb[n * 4 + 1] = y0 + lvl_shift;
        nmsb[n * 4 + 2] = x1 + lvl_shift; nmsb[n * 4 + 3] = y1 + lvl_shift;
        scores[n] = si;
    }
}

// ---------------- Stage 3: greedy NMS, one workgroup per image ----------------
__global__ __launch_bounds__(1024)
void nms_kernel(const float* __restrict__ boxes,
                const float* __restrict__ nmsb,
                const float* __restrict__ scores,
                float* __restrict__ out) {            // [B][1000][5]
    __shared__ float ssc[CATN];
    __shared__ float rs[32];
    __shared__ int   ri[32];
    __shared__ float4 wbox_s;
    __shared__ int    widx_s;

    const int b = blockIdx.x, tid = threadIdx.x;
    const int lane = tid & 31, wv = tid >> 5;

    for (int i = tid; i < CATN; i += 1024) ssc[i] = scores[(long)b * CATN + i];

    // cache this thread's NMS boxes in registers (boxes are immutable)
    float4 myb[5];
    {
        int c = 0;
        const float4* nb = (const float4*)(nmsb + (long)b * CATN * 4);
        for (int i = tid; i < CATN; i += 1024) myb[c++] = nb[i];
    }
    __syncthreads();

    for (int it = 0; it < POST_NMS; ++it) {
        // local argmax, first-index tie-break to match jnp.argmax
        float bs = -3.4e38f; int bi = 0x7fffffff;
        for (int i = tid; i < CATN; i += 1024) {
            const float s = ssc[i];
            if (s > bs || (s == bs && i < bi)) { bs = s; bi = i; }
        }
        // wave32 shuffle reduction (no barriers)
        #pragma unroll
        for (int m = 1; m < 32; m <<= 1) {
            const float s2 = __shfl_xor(bs, m, 32);
            const int   i2 = __shfl_xor(bi, m, 32);
            if (s2 > bs || (s2 == bs && i2 < bi)) { bs = s2; bi = i2; }
        }
        if (lane == 0) { rs[wv] = bs; ri[wv] = bi; }
        __syncthreads();
        if (wv == 0) {
            float s  = rs[lane];
            int   idx = ri[lane];
            #pragma unroll
            for (int m = 1; m < 32; m <<= 1) {
                const float s2 = __shfl_xor(s, m, 32);
                const int   i2 = __shfl_xor(idx, m, 32);
                if (s2 > s || (s2 == s && i2 < idx)) { s = s2; idx = i2; }
            }
            if (lane == 0) {
                widx_s = idx;
                const float4* nb = (const float4*)(nmsb + (long)b * CATN * 4);
                wbox_s = nb[idx];
                float* o = out + ((long)b * POST_NMS + it) * 5;
                if (s > NEGV * 0.5f) {
                    const float* bx = boxes + ((long)b * CATN + idx) * 4;
                    o[0] = bx[0]; o[1] = bx[1]; o[2] = bx[2]; o[3] = bx[3]; o[4] = s;
                } else {
                    o[0] = 0.f; o[1] = 0.f; o[2] = 0.f; o[3] = 0.f; o[4] = NEGV;
                }
            }
        }
        __syncthreads();
        const float4 wb = wbox_s;
        const int    w  = widx_s;
        const float  wa = (wb.z - wb.x) * (wb.w - wb.y);
        int c = 0;
        for (int i = tid; i < CATN; i += 1024) {
            const float4 bb = myb[c++];
            const float lx = fmaxf(wb.x, bb.x), ly = fmaxf(wb.y, bb.y);
            const float rx = fminf(wb.z, bb.z), ry = fminf(wb.w, bb.w);
            const float iw = fmaxf(rx - lx, 0.f), ih = fmaxf(ry - ly, 0.f);
            const float inter = iw * ih;
            const float ua = wa + (bb.z - bb.x) * (bb.w - bb.y) - inter;
            const float iou = inter / fmaxf(ua, 1e-9f);
            if (iou > 0.7f || i == w) ssc[i] = NEGV;
        }
        __syncthreads();
    }
}

// ---------------- Host-side orchestration ----------------
extern "C" void kernel_launch(void* const* d_in, const int* in_sizes, int n_in,
                              void* d_out, int out_size, void* d_ws, size_t ws_size,
                              hipStream_t stream) {
    static const int Hs[5] = {256, 128, 64, 32, 16};
    static const int Ws[5] = {256, 128, 64, 32, 16};
    static const int STR[5] = {4, 8, 16, 32, 64};
    static const int pixOff[5] = {0, 65536, 81920, 86016, 87040};  // prefix of HW
    static const int lvlOff[5] = {0, 1000, 2000, 3000, 4000};
    const int B = 2;
    const long totPix = 87296;

    const float* conv_b  = (const float*)d_in[6];
    const float* obj_w   = (const float*)d_in[7];
    const float* obj_b   = (const float*)d_in[8];
    const float* delta_w = (const float*)d_in[9];
    const float* delta_b = (const float*)d_in[10];

    char* ws = (char*)d_ws;
    unsigned short* inx   = (unsigned short*)ws;                       // 89,391,104 B
    size_t off = (size_t)B * totPix * 256 * 2;
    unsigned short* wpack = (unsigned short*)(ws + off); off += 72l * 16 * 512 * 2;
    float* logits = (float*)(ws + off); off += (size_t)B * totPix * 4;
    float* deltas = (float*)(ws + off); off += (size_t)B * totPix * 4 * 4;
    float* boxes  = (float*)(ws + off); off += (size_t)B * CATN * 4 * 4;
    float* nmsb   = (float*)(ws + off); off += (size_t)B * CATN * 4 * 4;
    float* scores = (float*)(ws + off); off += (size_t)B * CATN * 4;
    (void)ws_size;

    // Stage 0a: per-level NHWC bf16 conversion
    for (int l = 0; l < 5; ++l) {
        const int HW = Hs[l] * Ws[l];
        const long n = (long)B * HW * 256;
        unsigned short* dst = inx + (long)pixOff[l] * B * 256;
        convert_nhwc<<<dim3((unsigned)((n + 255) / 256)), 256, 0, stream>>>(
            (const float*)d_in[l], dst, HW, n);
    }

    // Stage 0b: weight packing (589,824 elements)
    pack_weights<<<dim3(2304), 256, 0, stream>>>((const float*)d_in[5], wpack);

    // Stage 1: fused conv + heads (WMMA)
    for (int l = 0; l < 5; ++l) {
        const int HW = Hs[l] * Ws[l];
        conv_rpn_head<<<dim3(HW / 128, B), 256, 0, stream>>>(
            inx + (long)pixOff[l] * B * 256, wpack,
            conv_b, obj_w, obj_b, delta_w, delta_b,
            logits + (long)pixOff[l] * B,
            deltas + (long)pixOff[l] * B * 4,
            Hs[l], Ws[l]);
    }

    // Stage 2: top-k + decode into concatenated arrays
    for (int l = 0; l < 5; ++l) {
        const int HW = Hs[l] * Ws[l];
        const int k = HW < 1000 ? HW : 1000;
        topk_decode<<<dim3(HW / 256, B), 256, 0, stream>>>(
            logits + (long)pixOff[l] * B,
            deltas + (long)pixOff[l] * B * 4,
            boxes, nmsb, scores,
            Hs[l], Ws[l], STR[l], k, lvlOff[l], (float)l * 1025.0f);
    }

    // Stage 3: NMS
    nms_kernel<<<dim3(B), 1024, 0, stream>>>(boxes, nmsb, scores, (float*)d_out);
}
